// PointwiseAggregatedAttention_1073741824700
// MI455X (gfx1250) — matrix operand order repaired and checked by reading, not compiled, not run on hardware
//
#include <hip/hip_runtime.h>
#include <hip/hip_bf16.h>

// ---------------------------------------------------------------------------
// Sigmoid (pointwise) multi-head attention, MI455X / gfx1250, wave32 WMMA.
//   B=2, S=4096, D=512, H=8, Hd=64, scale = 1/sqrt(64) = 0.125
// Pipeline (all bf16 matmul inputs, f32 WMMA accumulation):
//   1. Q = (X Wq + bq) * scale      -> ws  [B,H,S,Hd]  bf16
//      K = (X Wk + bk)              -> ws  [B,H,Hd,S]  bf16  (TRANSPOSED)
//      V = (X Wv + bv)              -> ws  [B,H,S,Hd]  bf16
//   2. O = sigmoid(Q K^T) V  (streamed over key tiles, no S x S buffer)
//                                   -> ws  [B,S,D]     bf16
//   3. out = O Wo + bo              -> d_out f32
// ---------------------------------------------------------------------------

#define S_LEN  4096
#define DM     512
#define NH     8
#define HD     64

typedef __attribute__((ext_vector_type(16))) __bf16 v16bf;
typedef __attribute__((ext_vector_type(8)))  __bf16 v8bf;
typedef __attribute__((ext_vector_type(8)))  float  v8f;

// float -> bf16, round-to-nearest-even
__device__ __forceinline__ __bf16 f2bf(float f) {
  unsigned u = __builtin_bit_cast(unsigned, f);
  unsigned r = u + 0x7FFFu + ((u >> 16) & 1u);
  unsigned short h = (unsigned short)(r >> 16);
  return __builtin_bit_cast(__bf16, h);
}

__device__ __forceinline__ v16bf combine(v8bf lo, v8bf hi) {
  v16bf a;
#pragma unroll
  for (int i = 0; i < 8; ++i) { a[i] = lo[i]; a[i + 8] = hi[i]; }
  return a;
}

// ---- fragment loaders (ISA 7.12.2 layouts, wave32) -------------------------
// A (16x32 bf16): lane m = l&15, half = l>>4.  VGPR i holds K pair at
//   kbase(i) = half*8 + (i>>2)*16 + (i&3)*2
// => v[0..3] = 8 consecutive K at half*8, v[4..7] = 8 consecutive at half*8+16.
__device__ __forceinline__ v16bf load_a(const __bf16* base, int ld, int lane) {
  const int m = lane & 15, half = lane >> 4;
  const __bf16* p = base + (size_t)m * ld + half * 8;
  v8bf lo = *(const v8bf*)(p);
  v8bf hi = *(const v8bf*)(p + 16);
  return combine(lo, hi);
}

__device__ __forceinline__ v16bf load_a(const float* base, int ld, int lane) {
  const int m = lane & 15, half = lane >> 4;
  const float* p = base + (size_t)m * ld + half * 8;
  float4 a0 = *(const float4*)(p);
  float4 a1 = *(const float4*)(p + 4);
  float4 b0 = *(const float4*)(p + 16);
  float4 b1 = *(const float4*)(p + 20);
  v16bf a;
  a[0]  = f2bf(a0.x); a[1]  = f2bf(a0.y); a[2]  = f2bf(a0.z); a[3]  = f2bf(a0.w);
  a[4]  = f2bf(a1.x); a[5]  = f2bf(a1.y); a[6]  = f2bf(a1.z); a[7]  = f2bf(a1.w);
  a[8]  = f2bf(b0.x); a[9]  = f2bf(b0.y); a[10] = f2bf(b0.z); a[11] = f2bf(b0.w);
  a[12] = f2bf(b1.x); a[13] = f2bf(b1.y); a[14] = f2bf(b1.z); a[15] = f2bf(b1.w);
  return a;
}

// B (32x16): lane l holds column n = l&15; elements are K = half*8+s (s<8), 16+half*8+(s-8) (s>=8).
__device__ __forceinline__ v16bf load_b(const __bf16* base, int ld, int lane) {
  const __bf16* p = base + (lane & 15) + (size_t)((lane >> 4) * 8) * ld;
  v16bf b;
#pragma unroll
  for (int s = 0; s < 8; ++s) { b[s] = p[(size_t)s * ld]; b[s + 8] = p[(size_t)(16 + s) * ld]; }
  return b;
}

__device__ __forceinline__ v16bf load_b(const float* base, int ld, int lane) {
  const float* p = base + (lane & 15) + (size_t)((lane >> 4) * 8) * ld;
  v16bf b;
#pragma unroll
  for (int s = 0; s < 8; ++s) { b[s] = f2bf(p[(size_t)s * ld]); b[s + 8] = f2bf(p[(size_t)(16 + s) * ld]); }
  return b;
}

__device__ __forceinline__ v8f wmma_bf16(v16bf a, v16bf b, v8f c) {
  // (neg_a, A, neg_b, B, c_mod, C, reuse_a, reuse_b)
  return __builtin_amdgcn_wmma_f32_16x16x32_bf16(false, a, false, b,
                                                 (short)0, c, false, false);
}

// ---------------------------------------------------------------------------
// GEMM:  Y[8192 x 512] = A[8192 x 512] * W[512 x 512] + bias
// Block = 64x64 output tile, 4 waves (128 thr), wave w -> rows [w*16, w*16+16).
// mode 0: Q  -> bf16 [B,H,S,Hd], scaled by 0.125
// mode 1: K  -> bf16 [B,H,Hd,S]  (transposed for contiguous QK^T B-frags)
// mode 2: V  -> bf16 [B,H,S,Hd]
// mode 3: final -> f32 [B,S,D] (d_out)
// ---------------------------------------------------------------------------
template <typename AT>
__global__ __launch_bounds__(128)
void proj_gemm_kernel(const AT* __restrict__ A, const float* __restrict__ W,
                      const float* __restrict__ bias,
                      __bf16* __restrict__ outb, float* __restrict__ outf,
                      int mode)
{
  const int lane = threadIdx.x & 31;
  const int wave = threadIdx.x >> 5;
  const int row0 = blockIdx.x * 64 + wave * 16;   // 8192 / 64 = 128 blocks in x
  const int col0 = blockIdx.y * 64;               // 512 / 64  = 8 blocks in y

  v8f acc[4] = {};
  for (int kk = 0; kk < DM; kk += 32) {
    v16bf a = load_a(A + (size_t)row0 * DM + kk, DM, lane);
#pragma unroll
    for (int t = 0; t < 4; ++t) {
      v16bf b = load_b(W + (size_t)kk * DM + col0 + t * 16, DM, lane);
      acc[t] = wmma_bf16(a, b, acc[t]);
    }
  }

  const int half = lane >> 4, ln = lane & 15;
#pragma unroll
  for (int t = 0; t < 4; ++t) {
    const int c  = col0 + t * 16 + ln;
    const float bv = bias[c];
#pragma unroll
    for (int r = 0; r < 8; ++r) {
      const int n = row0 + r + 8 * half;         // global row (b*S + s)
      float y = acc[t][r] + bv;
      if (mode == 3) {
        outf[(size_t)n * DM + c] = y;
      } else {
        const int b = n >> 12, s = n & (S_LEN - 1);
        const int h = c >> 6,  d = c & (HD - 1);
        if (mode == 0)       // Q, pre-scaled
          outb[(((size_t)(b * NH + h)) * S_LEN + s) * HD + d] = f2bf(y * 0.125f);
        else if (mode == 1)  // K transposed: [B,H,Hd,S]
          outb[(((size_t)(b * NH + h)) * HD + d) * S_LEN + s] = f2bf(y);
        else                 // V
          outb[(((size_t)(b * NH + h)) * S_LEN + s) * HD + d] = f2bf(y);
      }
    }
  }
}

// ---------------------------------------------------------------------------
// O[b,h,q,:] = sum_k sigmoid(Q[q,:] . K[k,:]) * V[k,:]   (scale folded into Q)
// Block = (b,h) x 64 query rows; wave w owns 16 rows, all 64 Hd columns.
// Streams keys in tiles of 64; P round-trips through a per-wave LDS slab to
// convert C-fragment layout -> A-fragment layout (same-wave DS ops are
// in-order, so no barrier needed).
// ---------------------------------------------------------------------------
__global__ __launch_bounds__(128)
void sigmoid_attn_kernel(const __bf16* __restrict__ Q,
                         const __bf16* __restrict__ KT,
                         const __bf16* __restrict__ V,
                         __bf16* __restrict__ O)
{
  __shared__ __align__(16) __bf16 Plds[4][16][80];   // stride 80 keeps b128 16B-aligned
  const int lane = threadIdx.x & 31;
  const int wave = threadIdx.x >> 5;
  const int q0 = blockIdx.x * 64 + wave * 16;  // 4096/64 = 64 blocks in x
  const int bh = blockIdx.y;                   // B*H = 16 blocks in y

  const __bf16* Qp = Q  + (size_t)bh * S_LEN * HD;   // [S, Hd]
  const __bf16* Kp = KT + (size_t)bh * HD * S_LEN;   // [Hd, S]
  const __bf16* Vp = V  + (size_t)bh * S_LEN * HD;   // [S, Hd]

  // Q A-fragments live in registers for the whole kernel (Hd = 64 = 2 frags)
  const v16bf aQ0 = load_a(Qp + (size_t)q0 * HD,      HD, lane);
  const v16bf aQ1 = load_a(Qp + (size_t)q0 * HD + 32, HD, lane);

  v8f oacc[4] = {};
  const int half = lane >> 4, ln = lane & 15;

  for (int j = 0; j < S_LEN; j += 64) {
    // ---- S = Q K^T over 4 key sub-tiles; sigmoid -> P (bf16) into LDS ----
#pragma unroll
    for (int t = 0; t < 4; ++t) {
      v16bf bK0 = load_b(Kp + (size_t)0  * S_LEN + j + t * 16, S_LEN, lane);
      v16bf bK1 = load_b(Kp + (size_t)32 * S_LEN + j + t * 16, S_LEN, lane);
      v8f s = {};
      s = wmma_bf16(aQ0, bK0, s);
      s = wmma_bf16(aQ1, bK1, s);
#pragma unroll
      for (int r = 0; r < 8; ++r) {
        float p = __builtin_amdgcn_rcpf(1.0f + __expf(-s[r]));
        Plds[wave][r + 8 * half][t * 16 + ln] = f2bf(p);
      }
    }

    // ---- re-read P as A-fragments (16 x 64 keys) ----
    const __bf16* pw = &Plds[wave][0][0];
    v16bf aP0 = load_a(pw,      80, lane);
    v16bf aP1 = load_a(pw + 32, 80, lane);

    // ---- O += P V over 4 Hd sub-tiles ----
#pragma unroll
    for (int t = 0; t < 4; ++t) {
      v16bf bV0 = load_b(Vp + (size_t)(j)      * HD + t * 16, HD, lane);
      v16bf bV1 = load_b(Vp + (size_t)(j + 32) * HD + t * 16, HD, lane);
      oacc[t] = wmma_bf16(aP0, bV0, oacc[t]);
      oacc[t] = wmma_bf16(aP1, bV1, oacc[t]);
    }

    // prefetch next key tile (global_prefetch_b8)
    if (j + 64 < S_LEN) {
      const int rr = threadIdx.x & 63;
      __builtin_prefetch(Kp + (size_t)rr * S_LEN + j + 64, 0, 1);
      __builtin_prefetch(Vp + (size_t)(j + 64 + rr) * HD, 0, 1);
    }
  }

  // ---- epilogue: write bf16 attn-out in [B, S, D] for the final GEMM ----
  const int b = bh >> 3, h = bh & 7;
#pragma unroll
  for (int t = 0; t < 4; ++t) {
#pragma unroll
    for (int r = 0; r < 8; ++r) {
      const int s = q0 + r + 8 * half;
      const int c = h * HD + t * 16 + ln;
      O[((size_t)b * S_LEN + s) * DM + c] = f2bf(oacc[t][r]);
    }
  }
}

// ---------------------------------------------------------------------------
extern "C" void kernel_launch(void* const* d_in, const int* in_sizes, int n_in,
                              void* d_out, int out_size, void* d_ws, size_t ws_size,
                              hipStream_t stream)
{
  const float* x  = (const float*)d_in[0];
  const float* Wq = (const float*)d_in[1];
  const float* bq = (const float*)d_in[2];
  const float* Wk = (const float*)d_in[3];
  const float* bk = (const float*)d_in[4];
  const float* Wv = (const float*)d_in[5];
  const float* bv = (const float*)d_in[6];
  const float* Wo = (const float*)d_in[7];
  const float* bo = (const float*)d_in[8];
  float* out = (float*)d_out;

  const size_t elems = (size_t)2 * NH * S_LEN * HD;  // 4M bf16 elements = 8 MB
  __bf16* qws = (__bf16*)d_ws;
  __bf16* kws = qws + elems;
  __bf16* vws = kws + elems;
  __bf16* aws = vws + elems;                          // 32 MB total workspace

  dim3 gp(128, 8), bp(128);
  proj_gemm_kernel<float><<<gp, bp, 0, stream>>>(x, Wq, bq, qws, nullptr, 0);
  proj_gemm_kernel<float><<<gp, bp, 0, stream>>>(x, Wk, bk, kws, nullptr, 1);
  proj_gemm_kernel<float><<<gp, bp, 0, stream>>>(x, Wv, bv, vws, nullptr, 2);
  sigmoid_attn_kernel<<<dim3(64, 16), bp, 0, stream>>>(qws, kws, vws, aws);
  proj_gemm_kernel<__bf16><<<gp, bp, 0, stream>>>(aws, Wo, bo, nullptr, out, 3);
}
